// Attention_18717467476095
// MI455X (gfx1250) — matrix operand order repaired and checked by reading, not compiled, not run on hardware
//
#include <hip/hip_runtime.h>

// ---------------------------------------------------------------------------
// Problem constants (match reference): B=8, S=2048, D_MODEL=1024, D_K=D_V=512
// ---------------------------------------------------------------------------
#define NB   8
#define SEQ  2048
#define DM   1024
#define DKV  512

typedef __attribute__((ext_vector_type(16))) __bf16 v16bf;
typedef __attribute__((ext_vector_type(8)))  float  v8f;

__device__ __forceinline__ unsigned short f2bf(float f) {
  unsigned int u = __float_as_uint(f);
  unsigned int r = (u + 0x7FFFu + ((u >> 16) & 1u)) >> 16;  // RNE
  return (unsigned short)r;
}

union FragU { uint4 u[2]; v16bf v; };

// A fragment (16x32 bf16): lane row = row; K = {kk+hi*8..+7, kk+16+hi*8..+7}
__device__ __forceinline__ v16bf load_frag_a(const unsigned short* base, int stride,
                                             int row, int kk, int hi) {
  FragU f;
  f.u[0] = *(const uint4*)(base + (size_t)row * stride + kk + hi * 8);
  f.u[1] = *(const uint4*)(base + (size_t)row * stride + kk + 16 + hi * 8);
  return f.v;
}
// B fragment (32x16 bf16), LDS holds B^T rows (n-major): K = {kk+hi*8..+7, kk+16+hi*8..+7}
__device__ __forceinline__ v16bf load_frag_b(const unsigned short* base, int stride,
                                             int n, int kk, int hi) {
  FragU f;
  f.u[0] = *(const uint4*)(base + (size_t)n * stride + kk + hi * 8);
  f.u[1] = *(const uint4*)(base + (size_t)n * stride + kk + 16 + hi * 8);
  return f.v;
}

__device__ __forceinline__ v8f wmma_bf16(v16bf a, v16bf b, v8f c) {
  return __builtin_amdgcn_wmma_f32_16x16x32_bf16(false, a, false, b, (short)0, c,
                                                 false, false);
}

// gfx1250 async copy: 16 bytes global -> LDS per lane, tracked by ASYNCcnt.
// LDS address = low 32 bits of the generic shared pointer (LDS aperture).
__device__ __forceinline__ void async_copy16(unsigned short* ldsDst,
                                             const unsigned short* gSrc) {
  unsigned lds = (unsigned)(unsigned long long)ldsDst;
  asm volatile("global_load_async_to_lds_b128 %0, %1, off"
               :: "v"(lds), "v"(gSrc) : "memory");
}
__device__ __forceinline__ void async_wait0() {
  asm volatile("s_wait_asynccnt 0x0" ::: "memory");
}

// ---------------------------------------------------------------------------
// Projection GEMM: Y[16384,512](bf16) = X[16384,1024](f32) @ W[1024,512](f32)
// Block tile 64x64, K-step 64. 8 waves = 4(M) x 2(N); each wave 16x32 C tile
// -> 4 WMMAs per stage, 2 barriers per 64 K.
// ---------------------------------------------------------------------------
__launch_bounds__(256)
__global__ void proj_gemm(const float* __restrict__ X, const float* __restrict__ W,
                          unsigned short* __restrict__ Y) {
  __shared__ unsigned short sX[64][72];  // 64x64 bf16 (+8 pad, 16B-aligned rows)
  __shared__ unsigned short sW[64][72];  // W^T tile: [n][k]

  const int mbase = blockIdx.x * 64;
  const int nbase = blockIdx.y * 64;
  const int t = threadIdx.x;
  const int w = t >> 5, lane = t & 31;
  const int wm = w >> 1, wn = w & 1;
  const int hi = lane >> 4, lr = lane & 15;

  v8f acc[2] = {{}, {}};
  for (int k0 = 0; k0 < DM; k0 += 64) {
    __syncthreads();
    // stage X tile 64x64 (f32 -> bf16): 1024 float4
#pragma unroll
    for (int i = 0; i < 4; ++i) {
      int idx = t + i * 256;
      int r = idx >> 4, c4 = idx & 15;
      float4 xv = *(const float4*)(X + (size_t)(mbase + r) * DM + k0 + c4 * 4);
      sX[r][c4 * 4 + 0] = f2bf(xv.x);
      sX[r][c4 * 4 + 1] = f2bf(xv.y);
      sX[r][c4 * 4 + 2] = f2bf(xv.z);
      sX[r][c4 * 4 + 3] = f2bf(xv.w);
    }
    // stage W tile transposed: read [k][n] coalesced, write sW[n][k]
#pragma unroll
    for (int i = 0; i < 4; ++i) {
      int idx = t + i * 256;
      int kr = idx >> 4, n4 = idx & 15;
      float4 wv = *(const float4*)(W + (size_t)(k0 + kr) * DKV + nbase + n4 * 4);
      sW[n4 * 4 + 0][kr] = f2bf(wv.x);
      sW[n4 * 4 + 1][kr] = f2bf(wv.y);
      sW[n4 * 4 + 2][kr] = f2bf(wv.z);
      sW[n4 * 4 + 3][kr] = f2bf(wv.w);
    }
    __syncthreads();
#pragma unroll
    for (int kk = 0; kk < 64; kk += 32) {
      v16bf a = load_frag_a(&sX[0][0], 72, wm * 16 + lr, kk, hi);
#pragma unroll
      for (int j = 0; j < 2; ++j) {
        v16bf b = load_frag_b(&sW[0][0], 72, wn * 32 + j * 16 + lr, kk, hi);
        acc[j] = wmma_bf16(a, b, acc[j]);
      }
    }
  }
#pragma unroll
  for (int j = 0; j < 2; ++j)
#pragma unroll
    for (int r = 0; r < 8; ++r) {
      int m = mbase + wm * 16 + r + 8 * hi;
      int n = nbase + wn * 32 + j * 16 + lr;
      Y[(size_t)m * DKV + n] = f2bf(acc[j][r]);
    }
}

// ---------------------------------------------------------------------------
// Flash attention, zero-redundancy WMMA partition.
// Block: 64 queries. 8 waves = 4 M-tiles x 2 halves.
//   QK^T : wave (mi,vi) computes S subtile [mi, key-half vi]  (16 WMMAs)
//   PV   : wave (mi,vi) accumulates O cols [vi*256, +256)     (16 WMMAs)
// Row softmax stats are merged across the two half-waves via LDS.
// Q/K chunks staged with GLOBAL_LOAD_ASYNC_TO_LDS_B128 (ASYNCcnt).
// ---------------------------------------------------------------------------
__launch_bounds__(256)
__global__ void flash_attn(const unsigned short* __restrict__ Qb,
                           const unsigned short* __restrict__ Kb,
                           const unsigned short* __restrict__ Vb,
                           const unsigned char* __restrict__ qpad,
                           const unsigned char* __restrict__ kpad,
                           float* __restrict__ Out) {
  __shared__ unsigned short sQ[64][520];   // 64 x 512 bf16, +8 pad
  __shared__ unsigned short sK[32][520];   // key chunk rows
  __shared__ unsigned short sVt[DKV][48];  // V^T chunk: [vcol][key], +16 pad
  __shared__ unsigned short sP[4][16][48]; // per-M-tile P (shared by half-waves)
  __shared__ float stMax[4][2][16];        // per-half row maxima
  __shared__ float stSum[4][2][16];        // per-half row exp-sums
  __shared__ unsigned char sKp[32];

  const int qt = blockIdx.x & 31;  // 32 query tiles (64 q) per batch
  const int b  = blockIdx.x >> 5;
  const int qbase = qt * 64;
  const int t = threadIdx.x, w = t >> 5, lane = t & 31;
  const int mi = w >> 1, vi = w & 1;
  const int hi = lane >> 4, lr = lane & 15;
  const size_t rowoff = (size_t)b * SEQ;
  const float scl = 0.044194173824159216f;  // 1/sqrt(512)

  // async-stage Q tile (64 x 512 bf16) once
#pragma unroll
  for (int i = 0; i < 16; ++i) {
    int idx = t + i * 256;  // 4096 x 16B
    int r = idx >> 6, c8 = idx & 63;
    async_copy16(&sQ[r][c8 * 8], Qb + ((rowoff + qbase + r) * DKV + c8 * 8));
  }
  int qdead[8];
#pragma unroll
  for (int r = 0; r < 8; ++r)
    qdead[r] = qpad[rowoff + qbase + mi * 16 + r + 8 * hi];

  float mrow[8], lsum[8];
  v8f o[16];
#pragma unroll
  for (int r = 0; r < 8; ++r) { mrow[r] = -1e30f; lsum[r] = 0.f; }
#pragma unroll
  for (int j = 0; j < 16; ++j) o[j] = (v8f){};

  const int nch = 2 * qt + 2;  // causal: keys 0 .. qbase+63
  for (int kc = 0; kc < nch; ++kc) {
    const int kbase = kc * 32;
    __syncthreads();
    // async-stage K chunk rows (32 x 512 bf16)
#pragma unroll
    for (int i = 0; i < 8; ++i) {
      int idx = t + i * 256;
      int r = idx >> 6, c8 = idx & 63;
      async_copy16(&sK[r][c8 * 8], Kb + ((rowoff + kbase + r) * DKV + c8 * 8));
    }
    // stage V chunk transposed: [key][vcol] -> sVt[vcol][key]
#pragma unroll
    for (int i = 0; i < 8; ++i) {
      int idx = t + i * 256;
      int r = idx >> 6, c8 = idx & 63;
      uint4 v4 = *(const uint4*)(Vb + ((rowoff + kbase + r) * DKV + c8 * 8));
      const unsigned short* e = (const unsigned short*)&v4;
#pragma unroll
      for (int j = 0; j < 8; ++j) sVt[c8 * 8 + j][r] = e[j];
    }
    if (t < 32) sKp[t] = kpad[rowoff + kbase + t];
    if (kc + 1 < nch)  // prefetch next V chunk (global_prefetch_b8)
      __builtin_prefetch(Vb + (rowoff + kbase + 32) * DKV, 0, 1);
    async_wait0();
    __syncthreads();

    // S subtile: 16 queries (mi) x 16 keys (half vi), over d_k = 512
    v8f s = {};
#pragma unroll
    for (int kk = 0; kk < DKV; kk += 32) {
      v16bf a  = load_frag_a(&sQ[0][0], 520, mi * 16 + lr, kk, hi);
      v16bf bb = load_frag_b(&sK[0][0], 520, vi * 16 + lr, kk, hi);
      s = wmma_bf16(a, bb, s);
    }

    // masks + local row maxima (row lives in one 16-lane half)
    const int key = kbase + vi * 16 + lr;
    const int kp = sKp[vi * 16 + lr];
    float v[8], rm8[8];
#pragma unroll
    for (int r = 0; r < 8; ++r) {
      int q = qbase + mi * 16 + r + 8 * hi;
      bool msk = qdead[r] || kp || (key > q);
      v[r] = msk ? -1e30f : s[r] * scl;
      float rm = v[r];
      rm = fmaxf(rm, __shfl_xor(rm, 1));
      rm = fmaxf(rm, __shfl_xor(rm, 2));
      rm = fmaxf(rm, __shfl_xor(rm, 4));
      rm = fmaxf(rm, __shfl_xor(rm, 8));
      rm8[r] = rm;
    }
    if (lr == 0) {
#pragma unroll
      for (int r = 0; r < 8; ++r) stMax[mi][vi][r + 8 * hi] = rm8[r];
    }
    __syncthreads();

    // merge maxima, compute P half, publish partial sums + P
    float ps8[8], mnew8[8];
#pragma unroll
    for (int r = 0; r < 8; ++r) {
      float other = stMax[mi][vi ^ 1][r + 8 * hi];
      float mnew = fmaxf(mrow[r], fmaxf(rm8[r], other));
      float p = (v[r] <= -1e29f) ? 0.f : __expf(v[r] - mnew);
      sP[mi][r + 8 * hi][vi * 16 + lr] = f2bf(p);
      float ps = p;
      ps += __shfl_xor(ps, 1);
      ps += __shfl_xor(ps, 2);
      ps += __shfl_xor(ps, 4);
      ps += __shfl_xor(ps, 8);
      ps8[r] = ps;
      mnew8[r] = mnew;
    }
    if (lr == 0) {
#pragma unroll
      for (int r = 0; r < 8; ++r) stSum[mi][vi][r + 8 * hi] = ps8[r];
    }
    __syncthreads();

    // finalize online-softmax state, rescale O
#pragma unroll
    for (int r = 0; r < 8; ++r) {
      float psTot = ps8[r] + stSum[mi][vi ^ 1][r + 8 * hi];
      float alpha = __expf(mrow[r] - mnew8[r]);  // dead rows: l,o stay 0
      lsum[r] = lsum[r] * alpha + psTot;
      mrow[r] = mnew8[r];
#pragma unroll
      for (int j = 0; j < 16; ++j) o[j][r] *= alpha;
    }

    // O += P @ V_chunk : cols vi*256 .. +255
    v16bf pa = load_frag_a(&sP[mi][0][0], 48, lr, 0, hi);
#pragma unroll
    for (int j = 0; j < 16; ++j) {
      v16bf bv = load_frag_b(&sVt[0][0], 48, vi * 256 + j * 16 + lr, 0, hi);
      o[j] = wmma_bf16(pa, bv, o[j]);
    }
  }

  // epilogue: out = O / l (0 when row fully masked)
#pragma unroll
  for (int r = 0; r < 8; ++r) {
    float inv = lsum[r] > 0.f ? 1.0f / lsum[r] : 0.f;
    int q = qbase + mi * 16 + r + 8 * hi;
#pragma unroll
    for (int j = 0; j < 16; ++j)
      Out[(rowoff + q) * DKV + vi * 256 + j * 16 + lr] = o[j][r] * inv;
  }
}

// ---------------------------------------------------------------------------
extern "C" void kernel_launch(void* const* d_in, const int* in_sizes, int n_in,
                              void* d_out, int out_size, void* d_ws, size_t ws_size,
                              hipStream_t stream) {
  (void)in_sizes; (void)n_in; (void)out_size; (void)ws_size;
  const float* sq  = (const float*)d_in[0];
  const float* skv = (const float*)d_in[1];
  const unsigned char* qpad = (const unsigned char*)d_in[2];
  const unsigned char* kpad = (const unsigned char*)d_in[3];
  const float* Wq = (const float*)d_in[4];
  const float* Wk = (const float*)d_in[5];
  const float* Wv = (const float*)d_in[6];
  float* out = (float*)d_out;

  // ws: Q,K,V projections in bf16: 3 * 8*2048*512*2 = 48 MB
  unsigned short* Qb = (unsigned short*)d_ws;
  unsigned short* Kb = Qb + (size_t)NB * SEQ * DKV;
  unsigned short* Vb = Kb + (size_t)NB * SEQ * DKV;

  dim3 pg((NB * SEQ) / 64, DKV / 64);
  proj_gemm<<<pg, 256, 0, stream>>>(sq,  Wq, Qb);
  proj_gemm<<<pg, 256, 0, stream>>>(skv, Wk, Kb);
  proj_gemm<<<pg, 256, 0, stream>>>(skv, Wv, Vb);

  flash_attn<<<NB * (SEQ / 64), 256, 0, stream>>>(Qb, Kb, Vb, qpad, kpad, out);
}
